// LinearSystem_8933531975721
// MI455X (gfx1250) — matrix operand order repaired and checked by reading, not compiled, not run on hardware
//
#include <hip/hip_runtime.h>

typedef __bf16 bf16_t;
typedef __attribute__((ext_vector_type(16))) __bf16 v16bf;
typedef __attribute__((ext_vector_type(8)))  __bf16 v8bf;
typedef __attribute__((ext_vector_type(8)))  float  v8f;

#define BATCH 64
#define TLEN  512
#define DIN   512
#define DOUT  256
#define SDIM  1024

// ---------------- conversion kernels (one-time) ----------------

__global__ void cvt_f32_bf16(const float* __restrict__ src, bf16_t* __restrict__ dst, int n) {
  int i = blockIdx.x * blockDim.x + threadIdx.x;
  if (i < n) dst[i] = (bf16_t)src[i];
}

// W [Nrows, Kcols] f32 row-major -> Wt [Kcols, Nrows] bf16
__global__ void cvt_transpose(const float* __restrict__ src, bf16_t* __restrict__ dst,
                              int Nrows, int Kcols) {
  int i = blockIdx.x * blockDim.x + threadIdx.x;
  if (i < Nrows * Kcols) {
    int n = i / Kcols, k = i % Kcols;
    dst[(size_t)k * Nrows + n] = (bf16_t)src[i];
  }
}

__global__ void zero_b32(unsigned* __restrict__ p, int n) {
  int i = blockIdx.x * blockDim.x + threadIdx.x;
  if (i < n) p[i] = 0u;
}

__device__ __forceinline__ v16bf load_b(const bf16_t* p, int N, int h) {
  v16bf b;
#pragma unroll
  for (int s = 0; s < 8; ++s) { b[s] = p[(size_t)(8 * h + s) * N]; b[s + 8] = p[(size_t)(16 + 8 * h + s) * N]; }
  return b;
}
// 4x v_wmma_f32_16x16x32_bf16 on one shared A fragment
__device__ __forceinline__ void wmma4(const v16bf& a, const v16bf& b0, const v16bf& b1,
                                      const v16bf& b2, const v16bf& b3, v8f acc[4]) {
  acc[0] = __builtin_amdgcn_wmma_f32_16x16x32_bf16(false, a, false, b0, (short)0, acc[0], false, false);
  acc[1] = __builtin_amdgcn_wmma_f32_16x16x32_bf16(false, a, false, b1, (short)0, acc[1], false, false);
  acc[2] = __builtin_amdgcn_wmma_f32_16x16x32_bf16(false, a, false, b2, (short)0, acc[2], false, false);
  acc[3] = __builtin_amdgcn_wmma_f32_16x16x32_bf16(false, a, false, b3, (short)0, acc[3], false, false);
}

// ---------------- big parallel GEMM: D = A(bf16)[M,K] * Bt(bf16)[K,N] ----------------
// One wave -> 16x64 tile. Explicit double-buffered pipeline: next step's A and B
// fragments are loaded into fresh registers before this step's 4 WMMAs issue, so
// the scheduler can use partial s_wait_loadcnt and hide L2 latency behind WMMAs.
// bu_mode: remap rows m=(b*TLEN+t) to time-major store [(t*BATCH+b)*N + n].
__global__ __launch_bounds__(256)
void wmma_gemm_bf16(const bf16_t* __restrict__ A, const bf16_t* __restrict__ Bt,
                    float* __restrict__ D, int M, int N, int K, int bu_mode) {
  const int lane = threadIdx.x & 31;
  const int wave = blockIdx.x * (blockDim.x >> 5) + (threadIdx.x >> 5);
  const int ngroups = N >> 6;
  const int mt = wave / ngroups;
  const int ng = wave % ngroups;
  if (mt * 16 >= M) return;                   // wave-uniform: EXEC stays all-1s
  const int m0 = mt * 16, n0 = ng * 64;

  const int hi   = lane >> 4;
  const int arow = m0 + (lane & 15);
  const int aoff = hi * 8;

  v8f acc[4] = {v8f{}, v8f{}, v8f{}, v8f{}};

  const bf16_t* aptr = A + (size_t)arow * K + aoff;
  const bf16_t* bptr = Bt + n0 + (lane & 15);   // B-frag: lane = n column, elements over k
  const size_t  bstep = (size_t)32 * N;

  // prologue: K-block 0 in registers
  v8bf  alo = *(const v8bf*)aptr;
  v8bf  ahi = *(const v8bf*)(aptr + 16);
  v16bf b0 = load_b(bptr, N, hi);
  v16bf b1 = load_b(bptr + 16, N, hi);
  v16bf b2 = load_b(bptr + 32, N, hi);
  v16bf b3 = load_b(bptr + 48, N, hi);

#pragma unroll 2
  for (int kk = 32; kk < K; kk += 32) {
    bptr += bstep;
    // issue next block's loads first (fresh registers -> partial waits)
    v16bf c0 = load_b(bptr, N, hi);
    v16bf c1 = load_b(bptr + 16, N, hi);
    v16bf c2 = load_b(bptr + 32, N, hi);
    v16bf c3 = load_b(bptr + 48, N, hi);
    v8bf  nlo = *(const v8bf*)(aptr + 32);
    v8bf  nhi = *(const v8bf*)(aptr + 48);
    aptr += 32;
    v16bf a;
#pragma unroll
    for (int i = 0; i < 8; ++i) { a[i] = alo[i]; a[i + 8] = ahi[i]; }
    wmma4(a, b0, b1, b2, b3, acc);
    alo = nlo; ahi = nhi;
    b0 = c0; b1 = c1; b2 = c2; b3 = c3;
  }
  { // epilogue: last K-block
    v16bf a;
#pragma unroll
    for (int i = 0; i < 8; ++i) { a[i] = alo[i]; a[i + 8] = ahi[i]; }
    wmma4(a, b0, b1, b2, b3, acc);
  }

  const int ncol  = lane & 15;
  const int rbase = hi * 8;
#pragma unroll
  for (int j = 0; j < 4; ++j)
#pragma unroll
    for (int r = 0; r < 8; ++r) {
      int m = m0 + rbase + r;
      int n = n0 + j * 16 + ncol;
      size_t idx = bu_mode
          ? ((size_t)(m & (TLEN - 1)) * BATCH + (m >> 9)) * (size_t)N + n
          : (size_t)m * N + n;
      D[idx] = acc[j][r];
    }
}

// ---------------- persistent scan kernel: ONE launch for all 512 steps ----------------
// 80 wave-tasks (10 blocks x 8 waves), co-resident; device-scope barrier per step.
// waves 0..63 : x_next[64,1024] = x_cur @ WaT + Bu_t (C-init)
// waves 64..79: out[b,t,:]      = x_cur @ WcT + Du   (C-init, already in d_out)
__global__ __launch_bounds__(256)
void scan_kernel(float* __restrict__ xa, float* __restrict__ xb,
                 const bf16_t* __restrict__ WaT, const bf16_t* __restrict__ WcT,
                 const float* __restrict__ Bu, float* __restrict__ out,
                 int* __restrict__ sync) {
  const int lane = threadIdx.x & 31;
  const int wave = blockIdx.x * (blockDim.x >> 5) + (threadIdx.x >> 5);
  const bool is_y = wave >= 64;
  const int  w    = is_y ? wave - 64 : wave;
  const int  ngroups = is_y ? (DOUT >> 6) : (SDIM >> 6);   // 4 | 16
  const int  N    = is_y ? DOUT : SDIM;
  const bf16_t* Bt = is_y ? WcT : WaT;
  const int m0 = (w / ngroups) * 16;
  const int n0 = (w % ngroups) * 64;

  const int hi    = lane >> 4;
  const int arow  = m0 + (lane & 15);
  const int aoff  = hi * 8;
  const int ncol  = lane & 15;
  const int rbase = hi * 8;

  const bf16_t* bbase = Bt + n0 + (lane & 15);
  const size_t  bstep = (size_t)32 * N;
  const size_t  arowoff = (size_t)arow * SDIM + aoff;
  const int nblocks = gridDim.x;

  for (int t = 0; t < TLEN; ++t) {
    const float* x_cur  = (t & 1) ? xb : xa;
    float*       x_next = (t & 1) ? xa : xb;
    const float* bu_t   = Bu + (size_t)t * BATCH * SDIM;

    // C-init from the input-driven terms (Bu_t for x-pass, Du for y-pass)
    v8f acc[4];
#pragma unroll
    for (int j = 0; j < 4; ++j)
#pragma unroll
      for (int r = 0; r < 8; ++r) {
        int m = m0 + rbase + r;
        int n = n0 + j * 16 + ncol;
        acc[j][r] = is_y ? out[((size_t)m * TLEN + t) * DOUT + n]
                         : bu_t[(size_t)m * SDIM + n];
      }

    const float*  xptr = x_cur + arowoff;
    const bf16_t* bptr = bbase;
    v8f   xlo = *(const v8f*)xptr;
    v8f   xhi = *(const v8f*)(xptr + 16);
    v16bf b0 = load_b(bptr, N, hi);
    v16bf b1 = load_b(bptr + 16, N, hi);
    v16bf b2 = load_b(bptr + 32, N, hi);
    v16bf b3 = load_b(bptr + 48, N, hi);

#pragma unroll 2
    for (int kk = 32; kk < SDIM; kk += 32) {
      bptr += bstep;
      v16bf c0 = load_b(bptr, N, hi);
      v16bf c1 = load_b(bptr + 16, N, hi);
      v16bf c2 = load_b(bptr + 32, N, hi);
      v16bf c3 = load_b(bptr + 48, N, hi);
      v8f   nlo = *(const v8f*)(xptr + 32);
      v8f   nhi = *(const v8f*)(xptr + 48);
      xptr += 32;
      v16bf a;
#pragma unroll
      for (int i = 0; i < 8; ++i) { a[i] = (bf16_t)xlo[i]; a[i + 8] = (bf16_t)xhi[i]; }
      wmma4(a, b0, b1, b2, b3, acc);
      xlo = nlo; xhi = nhi;
      b0 = c0; b1 = c1; b2 = c2; b3 = c3;
    }
    { // epilogue: last K-block
      v16bf a;
#pragma unroll
      for (int i = 0; i < 8; ++i) { a[i] = (bf16_t)xlo[i]; a[i + 8] = (bf16_t)xhi[i]; }
      wmma4(a, b0, b1, b2, b3, acc);
    }

#pragma unroll
    for (int j = 0; j < 4; ++j)
#pragma unroll
      for (int r = 0; r < 8; ++r) {
        int m = m0 + rbase + r;
        int n = n0 + j * 16 + ncol;
        if (is_y) out[((size_t)m * TLEN + t) * DOUT + n] = acc[j][r];
        else      x_next[(size_t)m * SDIM + n]           = acc[j][r];
      }

    // ---- device-scope grid barrier (monotonic counter, reset each call) ----
    __syncthreads();
    if (threadIdx.x == 0) {
      __hip_atomic_fetch_add(sync, 1, __ATOMIC_RELEASE, __HIP_MEMORY_SCOPE_AGENT);
      const int target = (t + 1) * nblocks;
      while (__hip_atomic_load(sync, __ATOMIC_ACQUIRE, __HIP_MEMORY_SCOPE_AGENT) < target) {
        __builtin_amdgcn_s_sleep(1);
      }
    }
    __syncthreads();
  }
}

// ---------------- launch ----------------

extern "C" void kernel_launch(void* const* d_in, const int* in_sizes, int n_in,
                              void* d_out, int out_size, void* d_ws, size_t ws_size,
                              hipStream_t stream) {
  const float* u  = (const float*)d_in[0];   // [64,512,512]
  const float* Wa = (const float*)d_in[1];   // [1024,1024]
  const float* Wb = (const float*)d_in[2];   // [1024,512]
  const float* Wc = (const float*)d_in[3];   // [256,1024]
  const float* Wd = (const float*)d_in[4];   // [256,512]
  float* out = (float*)d_out;                // [64,512,256]

  char* ws = (char*)d_ws;
  size_t off = 0;
  bf16_t* WaT = (bf16_t*)(ws + off); off += (size_t)SDIM * SDIM * 2;
  bf16_t* WbT = (bf16_t*)(ws + off); off += (size_t)DIN  * SDIM * 2;
  bf16_t* WcT = (bf16_t*)(ws + off); off += (size_t)SDIM * DOUT * 2;
  bf16_t* WdT = (bf16_t*)(ws + off); off += (size_t)DIN  * DOUT * 2;
  bf16_t* ubf = (bf16_t*)(ws + off); off += (size_t)BATCH * TLEN * DIN * 2;
  float*  Bu  = (float*)(ws + off);  off += (size_t)TLEN * BATCH * SDIM * 4; // [T][B][S]
  float*  x0  = (float*)(ws + off);  off += (size_t)BATCH * SDIM * 4;
  float*  x1  = (float*)(ws + off);  off += (size_t)BATCH * SDIM * 4;
  int*    sync = (int*)(ws + off);   off += 256;

  // one-time conversion + transposition; zero x0 and the barrier counter
  const int nu = BATCH * TLEN * DIN;
  cvt_f32_bf16 <<<(nu + 255) / 256, 256, 0, stream>>>(u, ubf, nu);
  cvt_transpose<<<(SDIM * SDIM + 255) / 256, 256, 0, stream>>>(Wa, WaT, SDIM, SDIM);
  cvt_transpose<<<(SDIM * DIN  + 255) / 256, 256, 0, stream>>>(Wb, WbT, SDIM, DIN);
  cvt_transpose<<<(DOUT * SDIM + 255) / 256, 256, 0, stream>>>(Wc, WcT, DOUT, SDIM);
  cvt_transpose<<<(DOUT * DIN  + 255) / 256, 256, 0, stream>>>(Wd, WdT, DOUT, DIN);
  zero_b32     <<<(BATCH * SDIM + 255) / 256, 256, 0, stream>>>((unsigned*)x0, BATCH * SDIM);
  zero_b32     <<<1, 64, 0, stream>>>((unsigned*)sync, 64);

  // Bu = u @ Wb^T  -> time-major [T][B][S]
  {
    int M = BATCH * TLEN, N = SDIM, K = DIN;
    int waves = (M / 16) * (N / 64);
    wmma_gemm_bf16<<<waves / 8, 256, 0, stream>>>(ubf, WbT, Bu, M, N, K, 1);
  }
  // Du = u @ Wd^T  -> straight into d_out (C-init for the y pass)
  {
    int M = BATCH * TLEN, N = DOUT, K = DIN;
    int waves = (M / 16) * (N / 64);
    wmma_gemm_bf16<<<waves / 8, 256, 0, stream>>>(ubf, WdT, out, M, N, K, 0);
  }
  // persistent scan: single launch, 512 internally-synchronized steps
  scan_kernel<<<10, 256, 0, stream>>>(x0, x1, WaT, WcT, Bu, out, sync);
}
